// WeightGenerator_18734647345367
// MI455X (gfx1250) — matrix operand order repaired and checked by reading, not compiled, not run on hardware
//
#include <hip/hip_runtime.h>
#include <hip/hip_bf16.h>
#include <math.h>

typedef __attribute__((ext_vector_type(16))) __bf16 v16bf;
typedef __attribute__((ext_vector_type(8)))  __bf16 v8bf;
typedef __attribute__((ext_vector_type(8)))  float  v8f;

#define B_  10
#define HI  64
#define WI  64
#define C_  64
#define M_  8
#define HB  128
#define WB  128
#define U_  64

// ---------------------------------------------------------------------------
// Transpose instance_feature [B,64,64,64,8] -> [B*8, 64,64,64].
// One pixel (512 contiguous floats) per block, staged in LDS with 9-float
// padded rows so both read and write phases are coalesced & conflict-free.
// ---------------------------------------------------------------------------
__global__ __launch_bounds__(256)
void transpose_inst_kernel(const float* __restrict__ in, float* __restrict__ out) {
    __shared__ float sb[64 * 9];                // [c*9 + m]
    const int pix = blockIdx.x;                 // over B*4096
    const float* src = in + (size_t)pix * 512;  // all (c,m) for this pixel
    const int t = threadIdx.x;

    #pragma unroll
    for (int i = 0; i < 2; ++i) {
        const int idx = t + i * 256;            // = c*8 + m (memory order)
        const int c = idx >> 3, m = idx & 7;
        sb[c * 9 + m] = src[idx];
    }
    __syncthreads();

    const int b = pix >> 12;
    const int p = pix & 4095;
    #pragma unroll
    for (int i = 0; i < 2; ++i) {
        const int idx = t + i * 256;            // = m*64 + c (output order)
        const int m = idx >> 6, c = idx & 63;
        out[((size_t)(b * 8 + m) * 4096 + p) * 64 + c] = sb[c * 9 + m];
    }
}

// ---------------------------------------------------------------------------
// Pre-convert conv1 weights [3,3,64,64] f32 (= [576][64]) to bf16 [64][576]
// (out-channel-major), so the conv kernel's sB stage is a pure vector copy.
// ---------------------------------------------------------------------------
__global__ __launch_bounds__(256)
void prep_w_kernel(const float* __restrict__ w, __bf16* __restrict__ wt) {
    const int idx = blockIdx.x * 256 + threadIdx.x;   // 576*64 = 36864
    if (idx >= 576 * 64) return;
    const int k = idx >> 6, n = idx & 63;
    wt[n * 576 + k] = (__bf16)w[idx];
}

// ---------------------------------------------------------------------------
// Conv1: 3x3, 64 -> 64 channels, SAME, + bias + ReLU. Implicit GEMM w/ WMMA
// bf16. Block = 256 threads (8 waves). Block tile: 128 pixels x 64 channels.
// K = 9*64 = 576, 18 steps of 32, double-buffered LDS (one barrier/step).
// ---------------------------------------------------------------------------
template<int H, int W>
__global__ __launch_bounds__(256)
void conv1_wmma_kernel(const float* __restrict__ in,    // [N,H,W,64] f32
                       const __bf16* __restrict__ wt,   // [64][576] bf16
                       const float* __restrict__ bias,  // [64]
                       __bf16* __restrict__ out)        // [N,H,W,64] bf16
{
    __shared__ __bf16 sA[2][128][40];   // 32 K + 8 pad (80B rows: conflict-free)
    __shared__ __bf16 sB[2][64][40];

    const int tid   = threadIdx.x;
    const int wave  = tid >> 5;
    const int lane  = tid & 31;
    const int half  = lane >> 4;
    const int rsel  = lane & 15;
    const int img   = blockIdx.y;
    const int Mbase = blockIdx.x * 128;
    const int row0  = wave * 16;
    const float* inimg = in + (size_t)img * H * W * 64;

    v8f acc[4];
    const v8f vzero = {0.f,0.f,0.f,0.f,0.f,0.f,0.f,0.f};
    #pragma unroll
    for (int t = 0; t < 4; ++t) acc[t] = vzero;

    // A-stage assignment: 2 threads per row, 16 K values each
    const int r_fill = tid >> 1;
    const int koff   = (tid & 1) * 16;
    const int p_fill = Mbase + r_fill;
    const int y_fill = p_fill / W;
    const int x_fill = p_fill % W;
    // B-stage assignment: 1 out-channel x 8 K values
    const int n_b  = tid >> 2;
    const int k0_b = (tid & 3) * 8;

    // ---- global fetch for step kk into registers (clamped addr, no OOB) ----
    float4 ga0, ga1, ga2, ga3;
    v8bf   gb;
    auto fetch = [&](int kk) {
        const int tap = kk >> 1;
        const int c0  = (kk & 1) * 32;
        const int dy  = tap / 3 - 1;
        const int dx  = tap % 3 - 1;
        int yy = y_fill + dy, xx = x_fill + dx;
        const bool ok = (yy >= 0) && (yy < H) && (xx >= 0) && (xx < W);
        yy = yy < 0 ? 0 : (yy > H - 1 ? H - 1 : yy);
        xx = xx < 0 ? 0 : (xx > W - 1 ? W - 1 : xx);
        const float4* gi = (const float4*)(inimg + ((size_t)(yy * W + xx) * 64 + c0 + koff));
        ga0 = gi[0]; ga1 = gi[1]; ga2 = gi[2]; ga3 = gi[3];
        if (!ok) {
            ga0 = make_float4(0.f, 0.f, 0.f, 0.f); ga1 = ga0; ga2 = ga0; ga3 = ga0;
        }
        gb = *(const v8bf*)(wt + n_b * 576 + kk * 32 + k0_b);
    };
    // ---- store staged registers into LDS buffer ----
    auto stage = [&](int buf) {
        v8bf p0, p1;
        p0[0]=(__bf16)ga0.x; p0[1]=(__bf16)ga0.y; p0[2]=(__bf16)ga0.z; p0[3]=(__bf16)ga0.w;
        p0[4]=(__bf16)ga1.x; p0[5]=(__bf16)ga1.y; p0[6]=(__bf16)ga1.z; p0[7]=(__bf16)ga1.w;
        p1[0]=(__bf16)ga2.x; p1[1]=(__bf16)ga2.y; p1[2]=(__bf16)ga2.z; p1[3]=(__bf16)ga2.w;
        p1[4]=(__bf16)ga3.x; p1[5]=(__bf16)ga3.y; p1[6]=(__bf16)ga3.z; p1[7]=(__bf16)ga3.w;
        *(v8bf*)&sA[buf][r_fill][koff]     = p0;
        *(v8bf*)&sA[buf][r_fill][koff + 8] = p1;
        *(v8bf*)&sB[buf][n_b][k0_b]        = gb;
    };

    // prologue: stage step 0 into buffer 0
    fetch(0);
    stage(0);
    __syncthreads();

    for (int kk = 0; kk < 18; ++kk) {
        const int cur = kk & 1;
        if (kk < 17) fetch(kk + 1);   // issue global loads early

        // A fragment: lane-half K split {0..7,16..23} / {8..15,24..31}
        const v8bf a0 = *(const v8bf*)&sA[cur][row0 + rsel][half * 8];
        const v8bf a1 = *(const v8bf*)&sA[cur][row0 + rsel][16 + half * 8];
        const v16bf afrag = __builtin_shufflevector(a0, a1,
            0,1,2,3,4,5,6,7,8,9,10,11,12,13,14,15);
        v16bf bfrag[4];
        #pragma unroll
        for (int nt = 0; nt < 4; ++nt) {
            const v8bf b0 = *(const v8bf*)&sB[cur][nt * 16 + rsel][half * 8];
            const v8bf b1 = *(const v8bf*)&sB[cur][nt * 16 + rsel][16 + half * 8];
            bfrag[nt] = __builtin_shufflevector(b0, b1,
                0,1,2,3,4,5,6,7,8,9,10,11,12,13,14,15);
        }
        if (kk < 17) stage(cur ^ 1);  // write other buffer (no hazard w/ reads)

        #pragma unroll
        for (int nt = 0; nt < 4; ++nt)
            acc[nt] = __builtin_amdgcn_wmma_f32_16x16x32_bf16(
                false, afrag, false, bfrag[nt], (short)0, acc[nt], false, false);

        if (kk < 17) __syncthreads();
    }

    // C/D layout: N = rsel (+nt*16), M = half*8 + j
    #pragma unroll
    for (int nt = 0; nt < 4; ++nt) {
        const int n  = nt * 16 + rsel;
        const float bv = bias[n];
        #pragma unroll
        for (int j = 0; j < 8; ++j) {
            const int p = Mbase + row0 + half * 8 + j;
            float v = acc[nt][j] + bv;
            v = v > 0.f ? v : 0.f;
            out[((size_t)img * H * W + p) * 64 + n] = (__bf16)v;
        }
    }
}

// ---------------------------------------------------------------------------
// Conv2: 3x3, 64 -> 1 channel, SAME, + bias + ReLU (dot of 576 per pixel)
// ---------------------------------------------------------------------------
template<int H, int W>
__global__ __launch_bounds__(256)
void conv2_kernel(const __bf16* __restrict__ in,   // [N,H,W,64] bf16
                  const float* __restrict__ wgt,   // [3,3,64,1] = [576]
                  const float* __restrict__ bias,  // [1]
                  float* __restrict__ out)         // [N,H,W]
{
    __shared__ float sw[576];
    for (int i = threadIdx.x; i < 576; i += 256) sw[i] = wgt[i];
    __syncthreads();

    const int img = blockIdx.y;
    const int p   = blockIdx.x * 256 + threadIdx.x;
    const int y   = p / W, x = p % W;
    const __bf16* inimg = in + (size_t)img * H * W * 64;

    float acc = 0.f;
    #pragma unroll
    for (int tap = 0; tap < 9; ++tap) {
        const int yy = y + tap / 3 - 1;
        const int xx = x + tap % 3 - 1;
        if (yy < 0 || yy >= H || xx < 0 || xx >= W) continue;
        const __bf16* pi = inimg + (size_t)(yy * W + xx) * 64;
        const float*  pw = &sw[tap * 64];
        #pragma unroll
        for (int c8 = 0; c8 < 8; ++c8) {
            const v8bf v = *(const v8bf*)(pi + c8 * 8);
            #pragma unroll
            for (int j = 0; j < 8; ++j)
                acc += (float)v[j] * pw[c8 * 8 + j];
        }
    }
    acc += bias[0];
    out[(size_t)img * H * W + p] = acc > 0.f ? acc : 0.f;
}

// ---------------------------------------------------------------------------
// Fused resize_and_pad + masked-exp softmax + composition.
// Block = 256 threads = 4 pixels x 64 channels. rfeat never materialized:
// bilinear-sample instance features directly (L2-resident, channel-contiguous).
// ---------------------------------------------------------------------------
__global__ __launch_bounds__(256)
void compose_kernel(const float* __restrict__ inst_t,   // [B*M,64,64,64]
                    const float* __restrict__ inst_w,   // [B*M,64,64]
                    const float* __restrict__ bg_w,     // [B,128,128]
                    const float* __restrict__ bg_feat,  // [B,128,128,64]
                    const float* __restrict__ bbox,     // [B,4,M]
                    const int*   __restrict__ object_n, // [B]
                    float* __restrict__ out)            // [B,128,128,64]
{
    const int c  = threadIdx.x & 63;
    const int pl = threadIdx.x >> 6;
    const int gp = blockIdx.x * 4 + pl;     // pixel over B*HB*WB
    const int b  = gp >> 14;
    const int pp = gp & 16383;
    const int hy = pp >> 7;
    const int hx = pp & 127;
    const int nobj = object_n[b];

    const float eb = expf(bg_w[gp]);
    float denom = eb;

    float eo[M_], fwy[M_], fwx[M_];
    int   iy0[M_], ix0[M_];

    #pragma unroll
    for (int m = 0; m < M_; ++m) {
        eo[m] = 0.f; fwy[m] = 0.f; fwx[m] = 0.f; iy0[m] = 0; ix0[m] = 0;
        if (m >= nobj) continue;
        const float by0 = bbox[(b * 4 + 0) * M_ + m];
        const float by1 = bbox[(b * 4 + 1) * M_ + m];
        const float bx0 = bbox[(b * 4 + 2) * M_ + m];
        const float bx1 = bbox[(b * 4 + 3) * M_ + m];
        const float ysz = floorf(128.f * (by1 - by0));
        const float xsz = floorf(128.f * (bx1 - bx0));
        const float yf  = floorf(128.f * by0);
        const float xf  = floorf(128.f * bx0);
        const float ly  = (float)hy - yf;
        const float lx  = (float)hx - xf;
        if (!((ly >= 0.f) && (ly < ysz) && (lx >= 0.f) && (lx < xsz))) continue;
        const float sy = fminf(fmaxf((ly + 0.5f) * 64.f / fmaxf(ysz, 1.f) - 0.5f, 0.f), 63.f);
        const float sx = fminf(fmaxf((lx + 0.5f) * 64.f / fmaxf(xsz, 1.f) - 0.5f, 0.f), 63.f);
        const int y0 = (int)floorf(sy);
        const int x0 = (int)floorf(sx);
        const int y1 = y0 + 1 < 63 ? y0 + 1 : 63;
        const int x1 = x0 + 1 < 63 ? x0 + 1 : 63;
        const float wy = sy - (float)y0;
        const float wx = sx - (float)x0;
        const float* fw = inst_w + (size_t)(b * M_ + m) * 4096;
        const float top = fw[y0 * 64 + x0] * (1.f - wx) + fw[y0 * 64 + x1] * wx;
        const float bot = fw[y1 * 64 + x0] * (1.f - wx) + fw[y1 * 64 + x1] * wx;
        const float rw  = top * (1.f - wy) + bot * wy;
        if (rw > 0.f) {
            eo[m]  = expf(rw);
            denom += eo[m];
            iy0[m] = y0; ix0[m] = x0; fwy[m] = wy; fwx[m] = wx;
        }
    }

    float res = (eb / denom) * bg_feat[(size_t)gp * 64 + c];

    #pragma unroll
    for (int m = 0; m < M_; ++m) {
        if (eo[m] == 0.f) continue;
        const int y0 = iy0[m], x0 = ix0[m];
        const int y1 = y0 + 1 < 63 ? y0 + 1 : 63;
        const int x1 = x0 + 1 < 63 ? x0 + 1 : 63;
        const float wy = fwy[m], wx = fwx[m];
        const float* ff = inst_t + (size_t)(b * M_ + m) * 4096 * 64;
        const float f00 = ff[(size_t)(y0 * 64 + x0) * 64 + c];
        const float f01 = ff[(size_t)(y0 * 64 + x1) * 64 + c];
        const float f10 = ff[(size_t)(y1 * 64 + x0) * 64 + c];
        const float f11 = ff[(size_t)(y1 * 64 + x1) * 64 + c];
        const float top = f00 * (1.f - wx) + f01 * wx;
        const float bot = f10 * (1.f - wx) + f11 * wx;
        res += (eo[m] / denom) * (top * (1.f - wy) + bot * wy);
    }
    out[(size_t)gp * 64 + c] = res;
}

// ---------------------------------------------------------------------------
extern "C" void kernel_launch(void* const* d_in, const int* in_sizes, int n_in,
                              void* d_out, int out_size, void* d_ws, size_t ws_size,
                              hipStream_t stream) {
    const float* instance_feature = (const float*)d_in[0];
    const float* bg_feature       = (const float*)d_in[1];
    const float* bbox             = (const float*)d_in[2];
    // d_in[3] mrcnn_mask: unused by the reference math
    const float* w1i = (const float*)d_in[4];
    const float* b1i = (const float*)d_in[5];
    const float* w2i = (const float*)d_in[6];
    const float* b2i = (const float*)d_in[7];
    const float* w1b = (const float*)d_in[8];
    const float* b1b = (const float*)d_in[9];
    const float* w2b = (const float*)d_in[10];
    const float* b2b = (const float*)d_in[11];
    const int* object_n = (const int*)d_in[12];

    // workspace carve-up (all offsets 256B-aligned)
    char* ws = (char*)d_ws;
    float*  inst_t  = (float*)ws;   ws += (size_t)80 * 4096 * 64 * 4;   // 83.9 MB
    __bf16* inst_h1 = (__bf16*)ws;  ws += (size_t)80 * 4096 * 64 * 2;   // 41.9 MB
    __bf16* bg_h1   = (__bf16*)ws;  ws += (size_t)10 * 16384 * 64 * 2;  // 21.0 MB
    float*  inst_w  = (float*)ws;   ws += (size_t)80 * 4096 * 4;        //  1.3 MB
    float*  bg_wv   = (float*)ws;   ws += (size_t)10 * 16384 * 4;       //  0.7 MB
    __bf16* w1i_t   = (__bf16*)ws;  ws += (size_t)64 * 576 * 2 + 256;   //  72 KB
    __bf16* w1b_t   = (__bf16*)ws;  ws += (size_t)64 * 576 * 2 + 256;   //  72 KB

    transpose_inst_kernel<<<40960, 256, 0, stream>>>(instance_feature, inst_t);

    prep_w_kernel<<<144, 256, 0, stream>>>(w1i, w1i_t);
    prep_w_kernel<<<144, 256, 0, stream>>>(w1b, w1b_t);

    conv1_wmma_kernel<64, 64><<<dim3(32, 80), 256, 0, stream>>>(inst_t, w1i_t, b1i, inst_h1);
    conv1_wmma_kernel<128, 128><<<dim3(128, 10), 256, 0, stream>>>(bg_feature, w1b_t, b1b, bg_h1);

    conv2_kernel<64, 64><<<dim3(16, 80), 256, 0, stream>>>(inst_h1, w2i, b2i, inst_w);
    conv2_kernel<128, 128><<<dim3(64, 10), 256, 0, stream>>>(bg_h1, w2b, b2b, bg_wv);

    compose_kernel<<<40960, 256, 0, stream>>>(inst_t, inst_w, bg_wv, bg_feature,
                                              bbox, object_n, (float*)d_out);
}
